// COVIDetector_13761075216703
// MI455X (gfx1250) — matrix operand order repaired and checked by reading, not compiled, not run on hardware
//
#include <hip/hip_runtime.h>
#include <math.h>
#include <stdint.h>

// ---------------------------------------------------------------------------
// CDNA5 / gfx1250 implementation of the COVIDetector reference:
//   CNN (3 fused conv+gelu+pool stages) -> 2 reversible Reformer blocks with
//   LSH attention -> CNN head -> linear.
// Dense matmuls: v_wmma_f32_16x16x32_f16 (wave32), LDS-staged via
// global_load_async_to_lds_* (ASYNCcnt) with an in-flight W transpose.
// ---------------------------------------------------------------------------

typedef __attribute__((ext_vector_type(16))) _Float16 v16h;
typedef __attribute__((ext_vector_type(8)))  _Float16 h8;
typedef __attribute__((ext_vector_type(8)))  float    v8f;
typedef __attribute__((ext_vector_type(4)))  float    f4;

#define HEADS     8
#define NHASH     4
#define BUCKET_SZ 64
#define DIMM      128
#define SEQT      512
#define BATCH     4
#define BHROWS    (BATCH * HEADS)            // 32
#define NHT       (NHASH * SEQT)             // 2048
#define NCHUNK    (NHASH * (SEQT/BUCKET_SZ)) // 32 chunks per bh row
#define ROWS      (BATCH * SEQT)             // 2048 sequence rows
#define SELF_VAL  (-5.0e4f)

__device__ __forceinline__ float gelu_f(float x) {
  return 0.5f * x * (1.0f + erff(x * 0.70710678118654752f));
}

__device__ __forceinline__ v8f wmma_16x16x32(v16h a, v16h b, v8f c) {
  return __builtin_amdgcn_wmma_f32_16x16x32_f16(false, a, false, b, (short)0, c,
                                                false, false);
}

// --- gfx1250 async global->LDS copies (tracked by ASYNCcnt). Generic LDS
// pointers carry the LDS byte offset in addr[31:0] (ISA 10.2), so the low
// 32 bits are exactly the LDS-relative address the instruction wants. ---
__device__ __forceinline__ void async_load_b128(void* lds, const void* g) {
  asm volatile("global_load_async_to_lds_b128 %0, %1, off"
               :: "v"((unsigned)(uintptr_t)lds),
                  "v"((unsigned long long)(uintptr_t)g)
               : "memory");
}
__device__ __forceinline__ void async_load_b32(void* lds, const void* g) {
  asm volatile("global_load_async_to_lds_b32 %0, %1, off"
               :: "v"((unsigned)(uintptr_t)lds),
                  "v"((unsigned long long)(uintptr_t)g)
               : "memory");
}
__device__ __forceinline__ void wait_async0() {
  asm volatile("s_wait_asynccnt 0" ::: "memory");
}

// ---------------------------------------------------------------------------
// Fused conv(2x2, stride2) + exact GELU + maxpool(2x2) kernel.
// IN_SEQ / OUT_SEQ: tensor stored (B, H*W, C) sequence-major.
// ---------------------------------------------------------------------------
template <int IN_SEQ, int OUT_SEQ>
__global__ void conv_gelu_pool_kernel(const float* __restrict__ in,
                                      const float* __restrict__ w,
                                      const float* __restrict__ bias,
                                      float* __restrict__ out,
                                      int B, int Cin, int Hin, int Win, int Cout) {
  int Hc = Hin >> 1, Wc = Win >> 1;
  int Hp = Hc >> 1,  Wp = Wc >> 1;
  int total = B * Cout * Hp * Wp;
  int idx = blockIdx.x * blockDim.x + threadIdx.x;
  if (idx >= total) return;
  int wp = idx % Wp; int t = idx / Wp;
  int hp = t % Hp;   t /= Hp;
  int co = t % Cout; int b = t / Cout;

  float best = -INFINITY;
  for (int py = 0; py < 2; ++py) {
    for (int px = 0; px < 2; ++px) {
      int hc = hp * 2 + py, wc = wp * 2 + px;
      float s = bias[co];
      for (int ci = 0; ci < Cin; ++ci) {
        const float* w4 = w + (co * Cin + ci) * 4;
        #pragma unroll
        for (int dy = 0; dy < 2; ++dy) {
          #pragma unroll
          for (int dx = 0; dx < 2; ++dx) {
            int hi = hc * 2 + dy, wi = wc * 2 + dx;
            float xin;
            if (IN_SEQ) xin = in[((size_t)b * (Hin * Win) + hi * Win + wi) * Cin + ci];
            else        xin = in[(((size_t)b * Cin + ci) * Hin + hi) * Win + wi];
            s += xin * w4[dy * 2 + dx];
          }
        }
      }
      best = fmaxf(best, gelu_f(s));
    }
  }
  if (OUT_SEQ) out[((size_t)b * (Hp * Wp) + hp * Wp + wp) * Cout + co] = best;
  else         out[(((size_t)b * Cout + co) * Hp + hp) * Wp + wp] = best;
}

// ---------------------------------------------------------------------------
// LayerNorm over last dim (=128).  One wave32 per row, shuffle reduction.
// ---------------------------------------------------------------------------
__global__ void layernorm_kernel(const float* __restrict__ in,
                                 const float* __restrict__ g,
                                 const float* __restrict__ b,
                                 float* __restrict__ out, int rows) {
  int row = blockIdx.x * (blockDim.x >> 5) + (threadIdx.x >> 5);
  int lane = threadIdx.x & 31;
  if (row >= rows) return;
  const float* p = in + (size_t)row * DIMM;
  float v[4]; float s = 0.0f;
  #pragma unroll
  for (int i = 0; i < 4; ++i) { v[i] = p[lane * 4 + i]; s += v[i]; }
  #pragma unroll
  for (int off = 16; off >= 1; off >>= 1) s += __shfl_xor(s, off, 32);
  float mu = s * (1.0f / 128.0f);
  float vs = 0.0f;
  #pragma unroll
  for (int i = 0; i < 4; ++i) { float d = v[i] - mu; vs += d * d; }
  #pragma unroll
  for (int off = 16; off >= 1; off >>= 1) vs += __shfl_xor(vs, off, 32);
  float inv = rsqrtf(vs * (1.0f / 128.0f) + 1e-5f);
  #pragma unroll
  for (int i = 0; i < 4; ++i) {
    int c = lane * 4 + i;
    out[(size_t)row * DIMM + c] = (v[i] - mu) * inv * g[c] + b[c];
  }
}

// ---------------------------------------------------------------------------
// WMMA GEMM:  C[M,N] = epilogue(A[M,K] @ W[K,N]).
// 32x64 macro-tile per block (8 waves, one 16x16 WMMA tile each).
// Per BK=32 step: A tile staged straight (async b128/lane), W tile staged
// TRANSPOSED to column-major [n][k] (async b32 scatter) so B-fragment reads
// are contiguous ds_load_b128 runs.  f32 LDS, f16 convert at fragment build.
// ---------------------------------------------------------------------------
template <int ACT, int HAS_BIAS, int HAS_RES>
__global__ __launch_bounds__(256)
void gemm_wmma_kernel(const float* __restrict__ A, const float* __restrict__ W,
                      const float* __restrict__ bias, const float* __restrict__ res,
                      float* __restrict__ C, int M, int N, int K) {
  __shared__ __align__(16) float s_a[32][32];   // [m][k]
  __shared__ __align__(16) float s_w[64][36];   // [n][k] (+4 pad: banks, 16B rows)

  int tid = threadIdx.x;
  int lane = tid & 31, wave = tid >> 5;
  int nbn = N >> 6;
  int bm = blockIdx.x / nbn, bn = blockIdx.x % nbn;
  int tm_l = wave >> 2, tn_l = wave & 3;

  int arow_l = tm_l * 16 + (lane & 15);        // local A row  (0..31)
  int bcol_l = tn_l * 16 + (lane & 15);        // local W col  (0..63)
  int kbase  = (lane < 16) ? 0 : 16;           // A-frag K half
  int kb0    = (lane < 16) ? 0 : 16;           // B-frag K half

  int a_row = tid >> 3, a_col = (tid & 7) * 4; // cooperative A copy map
  const float* Abase = A + ((size_t)(bm * 32 + a_row) * K + a_col);
  const float* Wbase = W + (size_t)bn * 64;

  v8f acc = {};
  for (int kk = 0; kk < K; kk += 32) {
    __syncthreads();                            // previous reads complete
    async_load_b128(&s_a[a_row][a_col], Abase + kk);
    #pragma unroll
    for (int i = 0; i < 8; ++i) {
      int idx = tid + i * 256;                  // 2048 dwords of W tile
      int k = idx >> 6, n = idx & 63;
      async_load_b32(&s_w[n][k], Wbase + (size_t)(kk + k) * N + n);
    }
    wait_async0();                              // own async copies done
    __syncthreads();                            // tile visible to all waves

    f4 a0 = *(const f4*)&s_a[arow_l][kbase];
    f4 a1 = *(const f4*)&s_a[arow_l][kbase + 4];
    f4 a2 = *(const f4*)&s_a[arow_l][kbase + 8];
    f4 a3 = *(const f4*)&s_a[arow_l][kbase + 12];
    f4 b0 = *(const f4*)&s_w[bcol_l][kb0];
    f4 b1 = *(const f4*)&s_w[bcol_l][kb0 + 4];
    f4 b2 = *(const f4*)&s_w[bcol_l][kb0 + 8];
    f4 b3 = *(const f4*)&s_w[bcol_l][kb0 + 12];

    v16h a, bf;
    #pragma unroll
    for (int h = 0; h < 4; ++h) {
      a[h]      = (_Float16)a0[h];
      a[4 + h]  = (_Float16)a1[h];
      a[8 + h]  = (_Float16)a2[h];
      a[12 + h] = (_Float16)a3[h];
      bf[h]      = (_Float16)b0[h];
      bf[4 + h]  = (_Float16)b1[h];
      bf[8 + h]  = (_Float16)b2[h];
      bf[12 + h] = (_Float16)b3[h];
    }
    acc = wmma_16x16x32(a, bf, acc);
  }

  int m0 = bm * 32 + tm_l * 16 + ((lane >> 4) << 3);
  int n  = bn * 64 + bcol_l;
  #pragma unroll
  for (int r = 0; r < 8; ++r) {
    int m = m0 + r;
    float vv = acc[r];
    if (HAS_BIAS) vv += bias[n];
    if (ACT == 1) vv = gelu_f(vv);
    if (HAS_RES)  vv += res[(size_t)m * N + n];
    C[(size_t)m * N + n] = vv;
  }
}

// ---------------------------------------------------------------------------
// LSH hashing: one thread per (bh, t).  rotated = qk_head(16) x rot(16,4,4);
// bucket = argmax over [rot, -rot] + hash*8.  key = (bucket*512+t)<<11 | j.
// ---------------------------------------------------------------------------
__global__ void lsh_hash_kernel(const float* __restrict__ qk,
                                const float* __restrict__ rot,
                                unsigned* __restrict__ keys) {
  int idx = blockIdx.x * blockDim.x + threadIdx.x;
  if (idx >= BHROWS * SEQT) return;
  int bh = idx >> 9, t = idx & 511;
  int b = bh >> 3, head = bh & 7;
  const float* q = qk + ((size_t)(b * SEQT + t) * DIMM + head * 16);
  float qv[16];
  #pragma unroll
  for (int f = 0; f < 16; ++f) qv[f] = q[f];

  for (int h = 0; h < NHASH; ++h) {
    float r4[4] = {0.f, 0.f, 0.f, 0.f};
    for (int f = 0; f < 16; ++f) {
      const float* rp = rot + (f * 16 + h * 4);  // rot[f][h][r]
      float qf = qv[f];
      r4[0] += qf * rp[0]; r4[1] += qf * rp[1];
      r4[2] += qf * rp[2]; r4[3] += qf * rp[3];
    }
    int best = 0; float bv = r4[0];
    #pragma unroll
    for (int r = 1; r < 8; ++r) {
      float val = (r < 4) ? r4[r] : -r4[r - 4];
      if (val > bv) { bv = val; best = r; }      // strict > keeps first tie
    }
    int bucket = best + h * 8;
    int j = h * SEQT + t;
    unsigned bt = (unsigned)(bucket * SEQT + t); // < 2^14
    keys[(size_t)bh * NHT + j] = (bt << 11) | (unsigned)j;
  }
}

// ---------------------------------------------------------------------------
// Bitonic sort of 2048 unique keys per (bh) row in LDS.
// ---------------------------------------------------------------------------
__global__ __launch_bounds__(1024)
void sort_kernel(const unsigned* __restrict__ keys, int* __restrict__ st,
                 int* __restrict__ undo) {
  __shared__ unsigned sk[NHT];
  int bh = blockIdx.x, tid = threadIdx.x;
  sk[tid]        = keys[(size_t)bh * NHT + tid];
  sk[tid + 1024] = keys[(size_t)bh * NHT + tid + 1024];
  __syncthreads();
  for (int k = 2; k <= NHT; k <<= 1) {
    for (int j = k >> 1; j > 0; j >>= 1) {
      #pragma unroll
      for (int e0 = 0; e0 < 2; ++e0) {
        int e = tid + e0 * 1024;
        int p = e ^ j;
        if (p > e) {
          bool asc = ((e & k) == 0);
          unsigned a = sk[e], b = sk[p];
          if ((a > b) == asc) { sk[e] = b; sk[p] = a; }
        }
      }
      __syncthreads();
    }
  }
  #pragma unroll
  for (int e0 = 0; e0 < 2; ++e0) {
    int pos = tid + e0 * 1024;
    int sidx = (int)(sk[pos] & 2047u);
    st[(size_t)bh * NHT + pos] = sidx & 511;   // original token index
    undo[(size_t)bh * NHT + sidx] = pos;       // inverse permutation
  }
}

// ---------------------------------------------------------------------------
// LSH chunk attention: one block per (bh, chunk), 64 queries x 128 keys,
// dh=16.  Q/K tiles stored zero-padded to K=32 so WMMA fragment builds are
// unconditional contiguous ds_load_b128 runs.  V gathered via async-to-LDS.
// ---------------------------------------------------------------------------
__global__ __launch_bounds__(128)
void lsh_attn_chunk_kernel(const float* __restrict__ qk, const float* __restrict__ v,
                           const int* __restrict__ st, float* __restrict__ so,
                           float* __restrict__ slog) {
  __shared__ int s_qt[64];
  __shared__ int s_kt[128];
  __shared__ __align__(16) _Float16 s_qpad[64][32];   // zero-padded K 16..31
  __shared__ __align__(16) _Float16 s_kpad[128][32];
  __shared__ __align__(16) float    s_v[128][16];
  __shared__ float s_d[64][128];

  int bid = blockIdx.x;
  int bh = bid >> 5, c = bid & 31;
  int b = bh >> 3, head = bh & 7;
  int tid = threadIdx.x;

  if (tid < 64) s_qt[tid] = st[(size_t)bh * NHT + c * 64 + tid];
  {
    int j = tid;                                     // 0..127 kv slots
    int cc = (j < 64) ? c : ((c + NCHUNK - 1) & 31); // look-one-back, wrap
    s_kt[j] = st[(size_t)bh * NHT + cc * 64 + (j & 63)];
  }
  __syncthreads();

  // --- async V gather: 128 rows x 64B, 4 x b128 per thread ---
  #pragma unroll
  for (int i = 0; i < 4; ++i) {
    int seg = tid + i * 128;
    int j = seg >> 2, e4 = (seg & 3) * 4;
    const float* gp = v + ((size_t)(b * SEQT + s_kt[j]) * DIMM + head * 16 + e4);
    async_load_b128(&s_v[j][e4], gp);
  }

  // --- synchronous K gather + L2 normalize, zero-padded to 32 ---
  {
    int j = tid, tok = s_kt[j];
    const float* kp = qk + ((size_t)(b * SEQT + tok) * DIMM + head * 16);
    float kr[16]; float nrm = 0.f;
    #pragma unroll
    for (int e = 0; e < 16; ++e) { kr[e] = kp[e]; nrm += kr[e] * kr[e]; }
    float inv = 1.0f / fmaxf(sqrtf(nrm), 1e-12f);
    #pragma unroll
    for (int e = 0; e < 16; ++e) {
      s_kpad[j][e]      = (_Float16)(kr[e] * inv);
      s_kpad[j][16 + e] = (_Float16)0.0f;
    }
  }
  if (tid < 64) {
    int tok = s_qt[tid];
    const float* qp = qk + ((size_t)(b * SEQT + tok) * DIMM + head * 16);
    #pragma unroll
    for (int e = 0; e < 16; ++e) {
      s_qpad[tid][e]      = (_Float16)qp[e];
      s_qpad[tid][16 + e] = (_Float16)0.0f;
    }
  }
  wait_async0();
  __syncthreads();

  // --- scores via WMMA: 4 waves x 8 N-tiles ---
  int wave = tid >> 5, lane = tid & 31;
  int qrow = wave * 16 + (lane & 15);
  int kbase = (lane < 16) ? 0 : 16;
  int kb0   = (lane < 16) ? 0 : 16;
  for (int nt = 0; nt < 8; ++nt) {
    h8 alo = *(const h8*)&s_qpad[qrow][kbase];
    h8 ahi = *(const h8*)&s_qpad[qrow][kbase + 8];
    h8 blo = *(const h8*)&s_kpad[nt * 16 + (lane & 15)][kb0];
    h8 bhi = *(const h8*)&s_kpad[nt * 16 + (lane & 15)][kb0 + 8];
    v16h a, bf;
    #pragma unroll
    for (int h = 0; h < 8; ++h) {
      a[h] = alo[h];  a[8 + h] = ahi[h];
      bf[h] = blo[h]; bf[8 + h] = bhi[h];
    }
    v8f acc = {};
    acc = wmma_16x16x32(a, bf, acc);
    #pragma unroll
    for (int r = 0; r < 8; ++r) {
      int i = wave * 16 + r + ((lane >> 4) << 3);
      int j = nt * 16 + (lane & 15);
      float d = acc[r] * 0.25f;                 // dh^-0.5 = 1/4
      if (s_qt[i] == s_kt[j]) d = SELF_VAL;     // shared-QK self mask
      s_d[i][j] = d;
    }
  }
  __syncthreads();

  // --- per-query logsumexp softmax + prob @ V ---
  if (tid < 64) {
    float m = -INFINITY;
    for (int j = 0; j < 128; ++j) m = fmaxf(m, s_d[tid][j]);
    float sum = 0.f;
    for (int j = 0; j < 128; ++j) sum += __expf(s_d[tid][j] - m);
    float lse = m + __logf(sum);
    float o[16];
    #pragma unroll
    for (int e = 0; e < 16; ++e) o[e] = 0.f;
    for (int j = 0; j < 128; ++j) {
      float p = __expf(s_d[tid][j] - lse);
      #pragma unroll
      for (int e = 0; e < 16; ++e) o[e] += p * s_v[j][e];
    }
    size_t gslot = (size_t)bh * NHT + c * 64 + tid;
    slog[gslot] = lse;
    #pragma unroll
    for (int e = 0; e < 16; ++e) so[gslot * 16 + e] = o[e];
  }
}

// ---------------------------------------------------------------------------
// Unsort + combine hash rounds (softmax over per-round lse), merge heads.
// ---------------------------------------------------------------------------
__global__ void combine_kernel(const float* __restrict__ so,
                               const float* __restrict__ slog,
                               const int* __restrict__ undo,
                               float* __restrict__ attnout) {
  int idx = blockIdx.x * blockDim.x + threadIdx.x;
  if (idx >= BATCH * SEQT * DIMM) return;
  int ccol = idx & 127;
  int t = (idx >> 7) & 511;
  int b = idx >> 16;
  int head = ccol >> 4, e = ccol & 15;
  int bh = b * HEADS + head;

  float lg[NHASH], val[NHASH]; float m = -INFINITY;
  #pragma unroll
  for (int h = 0; h < NHASH; ++h) {
    int p = undo[(size_t)bh * NHT + h * SEQT + t];
    lg[h]  = slog[(size_t)bh * NHT + p];
    val[h] = so[((size_t)bh * NHT + p) * 16 + e];
    m = fmaxf(m, lg[h]);
  }
  float den = 0.f, num = 0.f;
  #pragma unroll
  for (int h = 0; h < NHASH; ++h) {
    float w = __expf(lg[h] - m);
    den += w; num += w * val[h];
  }
  attnout[idx] = num / den;
}

__global__ void avg_kernel(const float* __restrict__ a, const float* __restrict__ b,
                           float* __restrict__ o, int n) {
  int i = blockIdx.x * blockDim.x + threadIdx.x;
  if (i < n) o[i] = 0.5f * (a[i] + b[i]);
}

__global__ void final_linear_kernel(const float* __restrict__ t,
                                    const float* __restrict__ lw,
                                    const float* __restrict__ lb,
                                    float* __restrict__ out) {
  int idx = threadIdx.x;
  if (idx >= 40) return;
  int b = idx / 10, o = idx % 10;
  out[idx] = t[b * 2 + 0] * lw[o] + t[b * 2 + 1] * lw[10 + o] + lb[o];
}

// ---------------------------------------------------------------------------
extern "C" void kernel_launch(void* const* d_in, const int* in_sizes, int n_in,
                              void* d_out, int out_size, void* d_ws, size_t ws_size,
                              hipStream_t stream) {
  (void)in_sizes; (void)n_in; (void)out_size; (void)ws_size;
  const float* x        = (const float*)d_in[0];
  const float* conv1_w  = (const float*)d_in[1];
  const float* conv1_b  = (const float*)d_in[2];
  const float* conv2_w  = (const float*)d_in[3];
  const float* conv2_b  = (const float*)d_in[4];
  const float* conv3_w  = (const float*)d_in[5];
  const float* conv3_b  = (const float*)d_in[6];
  const float* conv4_w  = (const float*)d_in[7];
  const float* conv4_b  = (const float*)d_in[8];
  const float* conv5_w  = (const float*)d_in[9];
  const float* conv5_b  = (const float*)d_in[10];
  const float* ln1_g    = (const float*)d_in[11];
  const float* ln1_b    = (const float*)d_in[12];
  const float* ln2_g    = (const float*)d_in[13];
  const float* ln2_b    = (const float*)d_in[14];
  const float* wqk      = (const float*)d_in[15];
  const float* wv       = (const float*)d_in[16];
  const float* wo       = (const float*)d_in[17];
  const float* wo_b     = (const float*)d_in[18];
  const float* ff_w1    = (const float*)d_in[19];
  const float* ff_b1    = (const float*)d_in[20];
  const float* ff_w2    = (const float*)d_in[21];
  const float* ff_b2    = (const float*)d_in[22];
  const float* rot      = (const float*)d_in[23];
  const float* lin_w    = (const float*)d_in[24];
  const float* lin_b    = (const float*)d_in[25];
  float* out = (float*)d_out;

  // ---- workspace layout (floats) ----
  float* ws = (float*)d_ws;
  size_t off = 0;
  auto alloc = [&](size_t n) { float* p = ws + off; off += n; return p; };
  float* c1      = alloc((size_t)4 * 32 * 256 * 512);
  float* c2      = alloc((size_t)4 * 64 * 64 * 128);
  float* seq     = alloc((size_t)ROWS * DIMM);
  float* bufA    = alloc((size_t)ROWS * DIMM);
  float* bufB    = alloc((size_t)ROWS * DIMM);
  float* bufC    = alloc((size_t)ROWS * DIMM);
  float* bufD    = alloc((size_t)ROWS * DIMM);
  float* lnbuf   = alloc((size_t)ROWS * DIMM);
  float* qkbuf   = alloc((size_t)ROWS * DIMM);
  float* vbuf    = alloc((size_t)ROWS * DIMM);
  float* attnout = alloc((size_t)ROWS * DIMM);
  float* ffmid   = alloc((size_t)ROWS * 512);
  float* so      = alloc((size_t)BHROWS * NHT * 16);
  float* slog    = alloc((size_t)BHROWS * NHT);
  unsigned* keys = (unsigned*)alloc((size_t)BHROWS * NHT);
  int* st        = (int*)alloc((size_t)BHROWS * NHT);
  int* undo      = (int*)alloc((size_t)BHROWS * NHT);
  float* c4      = alloc((size_t)4 * 64 * 4 * 8);
  float* c5      = alloc((size_t)4 * 2);

  // ---- CNN front-end ----
  {
    int n = 4 * 32 * 256 * 512;
    conv_gelu_pool_kernel<0, 0><<<(n + 255) / 256, 256, 0, stream>>>(
        x, conv1_w, conv1_b, c1, 4, 1, 1024, 2048, 32);
  }
  {
    int n = 4 * 64 * 64 * 128;
    conv_gelu_pool_kernel<0, 0><<<(n + 255) / 256, 256, 0, stream>>>(
        c1, conv2_w, conv2_b, c2, 4, 32, 256, 512, 64);
  }
  {
    int n = 4 * 128 * 16 * 32;
    conv_gelu_pool_kernel<0, 1><<<(n + 255) / 256, 256, 0, stream>>>(
        c2, conv3_w, conv3_b, seq, 4, 64, 64, 128, 128);
  }

  // ---- reversible Reformer blocks ----
  float* x1 = seq; float* x2 = seq;
  float* y1bufs[2] = {bufA, bufC};
  float* y2bufs[2] = {bufB, bufD};
  const int gemm128_blocks = (ROWS / 32) * (DIMM / 64);  // 128
  const int gemm512_blocks = (ROWS / 32) * (512 / 64);   // 512

  for (int d = 0; d < 2; ++d) {
    float* y1 = y1bufs[d];
    float* y2 = y2bufs[d];

    layernorm_kernel<<<ROWS / 8, 256, 0, stream>>>(x2, ln1_g + d * DIMM,
                                                   ln1_b + d * DIMM, lnbuf, ROWS);
    gemm_wmma_kernel<0, 0, 0><<<gemm128_blocks, 256, 0, stream>>>(
        lnbuf, wqk + (size_t)d * DIMM * DIMM, nullptr, nullptr, qkbuf, ROWS, DIMM, DIMM);
    gemm_wmma_kernel<0, 0, 0><<<gemm128_blocks, 256, 0, stream>>>(
        lnbuf, wv + (size_t)d * DIMM * DIMM, nullptr, nullptr, vbuf, ROWS, DIMM, DIMM);

    lsh_hash_kernel<<<(BHROWS * SEQT + 255) / 256, 256, 0, stream>>>(
        qkbuf, rot + (size_t)d * 16 * NHASH * 4, keys);
    sort_kernel<<<BHROWS, 1024, 0, stream>>>(keys, st, undo);
    lsh_attn_chunk_kernel<<<BHROWS * NCHUNK, 128, 0, stream>>>(qkbuf, vbuf, st, so, slog);
    combine_kernel<<<(BATCH * SEQT * DIMM + 255) / 256, 256, 0, stream>>>(
        so, slog, undo, attnout);

    // y1 = x1 + attnout @ wo + wo_b
    gemm_wmma_kernel<0, 1, 1><<<gemm128_blocks, 256, 0, stream>>>(
        attnout, wo + (size_t)d * DIMM * DIMM, wo_b + d * DIMM, x1, y1, ROWS, DIMM, DIMM);

    layernorm_kernel<<<ROWS / 8, 256, 0, stream>>>(y1, ln2_g + d * DIMM,
                                                   ln2_b + d * DIMM, lnbuf, ROWS);
    // ffmid = gelu(ln @ ff_w1 + ff_b1)
    gemm_wmma_kernel<1, 1, 0><<<gemm512_blocks, 256, 0, stream>>>(
        lnbuf, ff_w1 + (size_t)d * DIMM * 512, ff_b1 + d * 512, nullptr, ffmid,
        ROWS, 512, DIMM);
    // y2 = x2 + ffmid @ ff_w2 + ff_b2
    gemm_wmma_kernel<0, 1, 1><<<gemm128_blocks, 256, 0, stream>>>(
        ffmid, ff_w2 + (size_t)d * 512 * DIMM, ff_b2 + d * DIMM, x2, y2,
        ROWS, DIMM, 512);

    x1 = y1; x2 = y2;
  }

  // seq_final = 0.5 * (x1 + x2)
  avg_kernel<<<(ROWS * DIMM + 255) / 256, 256, 0, stream>>>(x1, x2, lnbuf, ROWS * DIMM);

  // ---- CNN head ----
  {
    int n = 4 * 64 * 4 * 8;
    conv_gelu_pool_kernel<1, 0><<<(n + 255) / 256, 256, 0, stream>>>(
        lnbuf, conv4_w, conv4_b, c4, 4, 128, 16, 32, 64);
  }
  {
    conv_gelu_pool_kernel<0, 0><<<1, 256, 0, stream>>>(
        c4, conv5_w, conv5_b, c5, 4, 64, 4, 8, 1);
  }
  final_linear_kernel<<<1, 64, 0, stream>>>(c5, lin_w, lin_b, out);
}
